// GQA_2310692405936
// MI455X (gfx1250) — matrix operand order repaired and checked by reading, not compiled, not run on hardware
//
#include <hip/hip_runtime.h>
#include <hip/hip_bf16.h>
#include <stdint.h>

// ---------- constants for this problem ----------
#define BB   4
#define TT   2048
#define CC   2048
#define NH   16
#define NKVH 4
#define HS   128                 // head size
#define KVSZ (NKVH * HS)         // 512
#define QKVW (CC + 2 * KVSZ)     // 3072

typedef __attribute__((ext_vector_type(16))) _Float16 v16h;
typedef __attribute__((ext_vector_type(8)))  _Float16 v8h;
typedef __attribute__((ext_vector_type(8)))  float    v8f;
typedef __attribute__((ext_vector_type(4)))  unsigned u32x4;
typedef __attribute__((ext_vector_type(8)))  int      i32x8;
typedef __attribute__((ext_vector_type(4)))  int      i32x4;

union U16h { v16h v; v8h h[2]; _Float16 e[16]; };

#if __has_builtin(__builtin_amdgcn_tensor_load_to_lds)
#define USE_TDM 1
#else
#define USE_TDM 0
#endif

// ---------------------------------------------------------------
// 1) f32 -> f16 elementwise convert
// ---------------------------------------------------------------
__global__ void conv_f32_to_f16(const float* __restrict__ src,
                                _Float16* __restrict__ dst, size_t n) {
    size_t i = (size_t)blockIdx.x * blockDim.x + threadIdx.x;
    if (i < n) dst[i] = (_Float16)src[i];
}

// ---------------------------------------------------------------
// 2) transpose-convert W [K][N] f32 -> Wt [N][K] f16
// ---------------------------------------------------------------
__global__ void transpose_f32_to_f16(const float* __restrict__ src,
                                     _Float16* __restrict__ dst,
                                     int K, int N) {
    size_t i = (size_t)blockIdx.x * blockDim.x + threadIdx.x;
    if (i >= (size_t)K * N) return;
    int    n = (int)(i % N);
    size_t k = i / N;
    dst[(size_t)n * K + k] = (_Float16)src[i];
}

#if USE_TDM
// ---------------------------------------------------------------
// Tensor Data Mover: DMA a [16 rows][32 halves] tile (row stride K
// elements) from global into LDS at lds_off, row-major, f16.
// 2D D# per CDNA5 ISA:
//   group0: count=1 | lds_addr | global_addr | type=2
//   group1: data_size=2B, tensor_dim0=K, tensor_dim1=16,
//           tile_dim0=32, tile_dim1=16, tensor_dim0_stride=K
// Groups 2/3 (and trailing group) zero: 2D tensor, no gather/iterate.
// Issued once per wave (TDM ignores EXEC); tracked by TENSORcnt.
// 6-arg builtin form (clang-23 / therock-10.0 toolchain).
// ---------------------------------------------------------------
__device__ __forceinline__ void tdm_load_16x32(const _Float16* gsrc,
                                               unsigned lds_off, int K) {
    unsigned long long ga = (unsigned long long)(uintptr_t)gsrc;
    unsigned galo = (unsigned)__builtin_amdgcn_readfirstlane((int)(unsigned)ga);
    unsigned gahi = (unsigned)__builtin_amdgcn_readfirstlane((int)(unsigned)(ga >> 32));
    unsigned loff = (unsigned)__builtin_amdgcn_readfirstlane((int)lds_off);

    u32x4 g0 = { 1u,                                   // count=1 (valid user D#)
                 loff,                                 // lds_addr
                 galo,                                 // global_addr[31:0]
                 (gahi & 0x01FFFFFFu) | 0x80000000u }; // global_addr[56:32] | type=2

    i32x8 g1 = { (int)0x00010000,                           // data_size=1 (2 bytes)
                 (int)((unsigned)(K & 0xFFFF) << 16),       // tensor_dim0[15:0]
                 (int)(((unsigned)(K >> 16) & 0xFFFFu) |    // tensor_dim0[31:16]
                       (16u << 16)),                        // | tensor_dim1[15:0]=16
                 (int)(32u << 16),                          // tile_dim0=32
                 (int)16,                                   // tile_dim1=16, tile_dim2=0
                 (int)K,                                    // tensor_dim0_stride[31:0]
                 0, 0 };

    i32x4 z4 = { 0, 0, 0, 0 };
    i32x8 z8 = { 0, 0, 0, 0, 0, 0, 0, 0 };
    __builtin_amdgcn_tensor_load_to_lds(g0, g1, z4, z4, z8, 0);
}
#endif

// ---------------------------------------------------------------
// 3) 128x128x32 WMMA GEMM:  C[M][N] = A[M][K] * Bt[N][K]^T + bias
//    256 threads = 8 waves; wave computes 64x32 (4x2 WMMA tiles).
//    TDM path: each wave DMAs its 16-row slice of the A/B tiles;
//    double-buffered LDS, DMA for step i+1 overlaps WMMA of step i.
// ---------------------------------------------------------------
__global__ __launch_bounds__(256)
void wmma_gemm_f16(const _Float16* __restrict__ A,
                   const _Float16* __restrict__ Bt,
                   const float* __restrict__ bias,
                   float* __restrict__ C,
                   int M, int N, int K) {
#if USE_TDM
    __shared__ _Float16 As[2][128][32];
    __shared__ _Float16 Bs[2][128][32];
#else
    __shared__ _Float16 As[1][128][32];
    __shared__ _Float16 Bs[1][128][32];
#endif

    const int tid  = threadIdx.x;
    const int lane = tid & 31;
    const int wid  = tid >> 5;
    const int waveM = (wid >> 2) * 64;   // 0 or 64
    const int waveN = (wid & 3)  * 32;   // 0,32,64,96
    const int bM = blockIdx.y * 128;
    const int bN = blockIdx.x * 128;

    v8f acc[4][2] = {};

#if USE_TDM
    const _Float16* Ap = A  + (size_t)(bM + wid * 16) * K;
    const _Float16* Bp = Bt + (size_t)(bN + wid * 16) * K;
    const unsigned aLds = (unsigned)(uintptr_t)&As[0][0][0] + (unsigned)wid * 1024u;
    const unsigned bLds = (unsigned)(uintptr_t)&Bs[0][0][0] + (unsigned)wid * 1024u;
    const int steps = K / 32;

    // prologue: DMA step 0 into buffer 0
    tdm_load_16x32(Ap, aLds, K);
    tdm_load_16x32(Bp, bLds, K);

    for (int it = 0; it < steps; ++it) {
        const int buf = it & 1;
        if (it + 1 < steps) {   // DMA next step into the other buffer
            unsigned nb = (unsigned)((it + 1) & 1) * 8192u;
            tdm_load_16x32(Ap + (size_t)(it + 1) * 32, aLds + nb, K);
            tdm_load_16x32(Bp + (size_t)(it + 1) * 32, bLds + nb, K);
            __builtin_amdgcn_s_wait_tensorcnt(2);   // in-order: current pair done
        } else {
            __builtin_amdgcn_s_wait_tensorcnt(0);
        }
        __syncthreads();        // all waves' slices of buf are resident
#else
    for (int k0 = 0; k0 < K; k0 += 32) {
        const int buf = 0;
        const int lr = tid >> 1;
        const int lc = (tid & 1) * 16;
        __syncthreads();
        *(v8h*)&As[0][lr][lc]     = *(const v8h*)&A [(size_t)(bM + lr) * K + k0 + lc];
        *(v8h*)&As[0][lr][lc + 8] = *(const v8h*)&A [(size_t)(bM + lr) * K + k0 + lc + 8];
        *(v8h*)&Bs[0][lr][lc]     = *(const v8h*)&Bt[(size_t)(bN + lr) * K + k0 + lc];
        *(v8h*)&Bs[0][lr][lc + 8] = *(const v8h*)&Bt[(size_t)(bN + lr) * K + k0 + lc + 8];
        __syncthreads();
#endif
        U16h bf[2];
        #pragma unroll
        for (int nt = 0; nt < 2; ++nt) {
            int n  = waveN + nt * 16 + (lane & 15);
            int kk = (lane >> 4) * 8;
            bf[nt].h[0] = *(const v8h*)&Bs[buf][n][kk];
            bf[nt].h[1] = *(const v8h*)&Bs[buf][n][kk + 16];
        }
        #pragma unroll
        for (int mt = 0; mt < 4; ++mt) {
            U16h af;
            int m  = waveM + mt * 16 + (lane & 15);
            int kk = (lane >> 4) * 8;
            af.h[0] = *(const v8h*)&As[buf][m][kk];
            af.h[1] = *(const v8h*)&As[buf][m][kk + 16];
            #pragma unroll
            for (int nt = 0; nt < 2; ++nt)
                acc[mt][nt] = __builtin_amdgcn_wmma_f32_16x16x32_f16(
                    false, af.v, false, bf[nt].v, (short)0, acc[mt][nt], false, false);
        }
#if USE_TDM
        __syncthreads();        // everyone done reading buf before it is re-DMAed
#endif
    }

    #pragma unroll
    for (int mt = 0; mt < 4; ++mt)
        #pragma unroll
        for (int nt = 0; nt < 2; ++nt) {
            int   gc = bN + waveN + nt * 16 + (lane & 15);
            float bv = bias ? bias[gc] : 0.0f;
            #pragma unroll
            for (int i = 0; i < 8; ++i) {
                int gr = bM + waveM + mt * 16 + i + ((lane >> 4) << 3);
                C[(size_t)gr * N + gc] = acc[mt][nt][i] + bv;
            }
        }
}

// ---------------------------------------------------------------
// 4) RoPE + split + relayout:
//    qkv f32 [B][T][3072] -> Qh f16 [B][NH][T][HS],
//                            Kh f16 [B][NKVH][T][HS],
//                            Vt f16 [B][NKVH][HS][T]
// ---------------------------------------------------------------
__global__ void rope_split(const float* __restrict__ qkv,
                           const float* __restrict__ fc,
                           const float* __restrict__ fs,
                           _Float16* __restrict__ Qh,
                           _Float16* __restrict__ Kh,
                           _Float16* __restrict__ Vt) {
    const int PAIRS = (NH + 2 * NKVH) * (HS / 2);   // 1536 per token
    size_t idx = (size_t)blockIdx.x * blockDim.x + threadIdx.x;
    if (idx >= (size_t)BB * TT * PAIRS) return;
    int    p   = (int)(idx % PAIRS);
    size_t tok = idx / PAIRS;
    int t = (int)(tok % TT);
    int b = (int)(tok / TT);
    const float* base = qkv + tok * QKVW;

    if (p < NH * (HS / 2)) {                         // ---- Q, rope
        int hh = p / (HS / 2), i = p % (HS / 2);
        float re = base[hh * HS + 2 * i];
        float im = base[hh * HS + 2 * i + 1];
        float c = fc[(size_t)t * (HS / 2) + i];
        float s = fs[(size_t)t * (HS / 2) + i];
        size_t o = ((size_t)(b * NH + hh) * TT + t) * HS + 2 * i;
        Qh[o]     = (_Float16)(re * c - im * s);
        Qh[o + 1] = (_Float16)(re * s + im * c);
    } else if (p < (NH + NKVH) * (HS / 2)) {         // ---- K, rope
        int p2 = p - NH * (HS / 2);
        int hh = p2 / (HS / 2), i = p2 % (HS / 2);
        float re = base[CC + hh * HS + 2 * i];
        float im = base[CC + hh * HS + 2 * i + 1];
        float c = fc[(size_t)t * (HS / 2) + i];
        float s = fs[(size_t)t * (HS / 2) + i];
        size_t o = ((size_t)(b * NKVH + hh) * TT + t) * HS + 2 * i;
        Kh[o]     = (_Float16)(re * c - im * s);
        Kh[o + 1] = (_Float16)(re * s + im * c);
    } else {                                         // ---- V, transpose to [HS][T]
        int p3 = p - (NH + NKVH) * (HS / 2);
        int hh = p3 / (HS / 2), i = p3 % (HS / 2);
        int d = 2 * i;
        float v0 = base[CC + KVSZ + hh * HS + d];
        float v1 = base[CC + KVSZ + hh * HS + d + 1];
        size_t o = ((size_t)(b * NKVH + hh) * HS + d) * TT + t;
        Vt[o]      = (_Float16)v0;
        Vt[o + TT] = (_Float16)v1;
    }
}

// ---------------------------------------------------------------
// 5) Flash attention (causal, GQA). 128 threads = 4 waves.
//    Each wave owns one 16-row Q tile; block covers 64 q rows.
//    Per 32-key chunk: 8 WMMA (S) + softmax + 8 WMMA (PV).
// ---------------------------------------------------------------
__global__ __launch_bounds__(128)
void flash_attn(const _Float16* __restrict__ Q,
                const _Float16* __restrict__ Km,
                const _Float16* __restrict__ Vt,
                _Float16* __restrict__ Y) {
    __shared__ _Float16 Pl[4][16][32];   // per-wave P staging tile

    const int lane = threadIdx.x & 31;
    const int wid  = threadIdx.x >> 5;
    const int qTiles = TT / 64;
    const int qt  = blockIdx.x % qTiles;
    const int bh  = blockIdx.x / qTiles;
    const int b   = bh / NH;
    const int h   = bh % NH;
    const int kvh = h / (NH / NKVH);

    const _Float16* Qp = Q  + ((size_t)(b * NH   + h)   * TT) * HS;
    const _Float16* Kp = Km + ((size_t)(b * NKVH + kvh) * TT) * HS;
    const _Float16* Vp = Vt + ((size_t)(b * NKVH + kvh) * HS) * TT;

    const int qRowBase = qt * 64 + wid * 16;
    const int qr = qRowBase + (lane & 15);

    // Q fragments (A layout), fixed for the whole KV loop
    U16h qf[4];
    #pragma unroll
    for (int c = 0; c < 4; ++c) {
        int d = c * 32 + ((lane >> 4) << 3);
        qf[c].h[0] = *(const v8h*)&Qp[(size_t)qr * HS + d];
        qf[c].h[1] = *(const v8h*)&Qp[(size_t)qr * HS + d + 16];
    }

    float m[8], l[8];
    v8f   O[8] = {};
    #pragma unroll
    for (int i = 0; i < 8; ++i) { m[i] = -3.0e38f; l[i] = 0.0f; }

    const float scale = 0.08838834764831845f;   // 1/sqrt(128)
    const int kEnd = qt * 64 + 64;              // causal bound for this block

    for (int kb = 0; kb < kEnd; kb += 32) {
        // ---- S = Q K^T for 32 keys (two 16x16 C tiles)
        v8f s[2] = {};
        #pragma unroll
        for (int nt = 0; nt < 2; ++nt) {
            int key = kb + nt * 16 + (lane & 15);
            #pragma unroll
            for (int c = 0; c < 4; ++c) {
                U16h bf;
                int d = c * 32 + ((lane >> 4) << 3);
                bf.h[0] = *(const v8h*)&Kp[(size_t)key * HS + d];
                bf.h[1] = *(const v8h*)&Kp[(size_t)key * HS + d + 16];
                s[nt] = __builtin_amdgcn_wmma_f32_16x16x32_f16(
                    false, qf[c].v, false, bf.v, (short)0, s[nt], false, false);
            }
        }

        // ---- online softmax update (rows split: lanes<16 -> rows i, lanes>=16 -> rows i+8)
        float sf[8];
        #pragma unroll
        for (int i = 0; i < 8; ++i) {
            int row = qRowBase + i + ((lane >> 4) << 3);
            int k0i = kb + (lane & 15);
            int k1i = kb + 16 + (lane & 15);
            float v0 = (k0i <= row) ? s[0][i] * scale : -3.0e38f;
            float v1 = (k1i <= row) ? s[1][i] * scale : -3.0e38f;
            float mx = fmaxf(v0, v1);
            #pragma unroll
            for (int off = 8; off >= 1; off >>= 1)
                mx = fmaxf(mx, __shfl_xor(mx, off, 16));
            float mnew = fmaxf(m[i], mx);
            sf[i] = __expf(m[i] - mnew);
            float p0 = (k0i <= row) ? __expf(v0 - mnew) : 0.0f;
            float p1 = (k1i <= row) ? __expf(v1 - mnew) : 0.0f;
            int prow = i + ((lane >> 4) << 3);
            Pl[wid][prow][lane & 15]        = (_Float16)p0;
            Pl[wid][prow][16 + (lane & 15)] = (_Float16)p1;
            float r = p0 + p1;
            #pragma unroll
            for (int off = 8; off >= 1; off >>= 1)
                r += __shfl_xor(r, off, 16);
            l[i] = l[i] * sf[i] + r;
            m[i] = mnew;
        }
        #pragma unroll
        for (int j = 0; j < 8; ++j)
            #pragma unroll
            for (int i = 0; i < 8; ++i) O[j][i] *= sf[i];

        // order in-wave LDS RAW (P written in C layout, re-read in A layout)
        asm volatile("s_wait_dscnt 0x0" ::: "memory");

        U16h pf;
        {
            int kk = (lane >> 4) << 3;
            pf.h[0] = *(const v8h*)&Pl[wid][lane & 15][kk];
            pf.h[1] = *(const v8h*)&Pl[wid][lane & 15][kk + 16];
        }

        // ---- O += P V  (reduction over this 32-key chunk)
        #pragma unroll
        for (int j = 0; j < 8; ++j) {
            U16h vf;
            int d  = j * 16 + (lane & 15);
            int kk = kb + ((lane >> 4) << 3);
            vf.h[0] = *(const v8h*)&Vp[(size_t)d * TT + kk];
            vf.h[1] = *(const v8h*)&Vp[(size_t)d * TT + kk + 16];
            O[j] = __builtin_amdgcn_wmma_f32_16x16x32_f16(
                false, pf.v, false, vf.v, (short)0, O[j], false, false);
        }
    }

    // ---- normalize and write y (f16, [B][T][C] layout for the proj GEMM)
    #pragma unroll
    for (int i = 0; i < 8; ++i) {
        float inv = 1.0f / l[i];
        int row = qRowBase + i + ((lane >> 4) << 3);
        #pragma unroll
        for (int j = 0; j < 8; ++j) {
            int d = h * HS + j * 16 + (lane & 15);
            Y[((size_t)b * TT + row) * CC + d] = (_Float16)(O[j][i] * inv);
        }
    }
}

// ---------------------------------------------------------------
// launcher
// ---------------------------------------------------------------
extern "C" void kernel_launch(void* const* d_in, const int* in_sizes, int n_in,
                              void* d_out, int out_size, void* d_ws, size_t ws_size,
                              hipStream_t stream) {
    const float* x  = (const float*)d_in[0];
    const float* fc = (const float*)d_in[1];
    const float* fs = (const float*)d_in[2];
    const float* Wa = (const float*)d_in[3];
    const float* ba = (const float*)d_in[4];
    const float* Wp = (const float*)d_in[5];
    const float* bp = (const float*)d_in[6];
    float* out = (float*)d_out;

    char* ws = (char*)d_ws;
    size_t off = 0;
    auto alloc = [&](size_t bytes) -> void* {
        void* p = ws + off;
        off += (bytes + 255) & ~(size_t)255;
        return p;
    };

    const size_t nX   = (size_t)BB * TT * CC;          // 16.8M
    const size_t nWa  = (size_t)CC * QKVW;             // 6.3M
    const size_t nWp  = (size_t)CC * CC;               // 4.2M
    const size_t nQKV = (size_t)BB * TT * QKVW;        // 25.2M

    _Float16* xh  = (_Float16*)alloc(nX * 2);
    _Float16* Wat = (_Float16*)alloc(nWa * 2);
    _Float16* Wpt = (_Float16*)alloc(nWp * 2);
    float*    qkv = (float*)   alloc(nQKV * 4);
    _Float16* Qh  = (_Float16*)alloc((size_t)BB * NH   * TT * HS * 2);
    _Float16* Kh  = (_Float16*)alloc((size_t)BB * NKVH * TT * HS * 2);
    _Float16* Vt  = (_Float16*)alloc((size_t)BB * NKVH * HS * TT * 2);
    _Float16* Yh  = (_Float16*)alloc(nX * 2);

    // stage 1: conversions
    conv_f32_to_f16<<<(unsigned)((nX + 255) / 256), 256, 0, stream>>>(x, xh, nX);
    transpose_f32_to_f16<<<(unsigned)((nWa + 255) / 256), 256, 0, stream>>>(Wa, Wat, CC, QKVW);
    transpose_f32_to_f16<<<(unsigned)((nWp + 255) / 256), 256, 0, stream>>>(Wp, Wpt, CC, CC);

    // stage 2: QKV GEMM  [8192 x 2048] * [2048 x 3072]
    {
        dim3 grid(QKVW / 128, (BB * TT) / 128);
        wmma_gemm_f16<<<grid, 256, 0, stream>>>(xh, Wat, ba, qkv, BB * TT, QKVW, CC);
    }

    // stage 3: RoPE + split + V transpose
    {
        size_t total = (size_t)BB * TT * (NH + 2 * NKVH) * (HS / 2);
        rope_split<<<(unsigned)((total + 255) / 256), 256, 0, stream>>>(qkv, fc, fs, Qh, Kh, Vt);
    }

    // stage 4: flash attention
    {
        unsigned blocks = (unsigned)(BB * NH * (TT / 64));
        flash_attn<<<blocks, 128, 0, stream>>>(Qh, Kh, Vt, Yh);
    }

    // stage 5: output projection  [8192 x 2048] * [2048 x 2048] + bias
    {
        dim3 grid(CC / 128, (BB * TT) / 128);
        wmma_gemm_f16<<<grid, 256, 0, stream>>>(Yh, Wpt, bp, out, BB * TT, CC, CC);
    }
}
